// Transformer_Encoder_66494683677361
// MI455X (gfx1250) — matrix operand order repaired and checked by reading, not compiled, not run on hardware
//
#include <hip/hip_runtime.h>
#include <hip/hip_bf16.h>
#include <math.h>

// ---------------------------------------------------------------------------
// Transformer-XL encoder on gfx1250 (MI455X), bf16 WMMA everywhere.
// ---------------------------------------------------------------------------

typedef __attribute__((ext_vector_type(16))) __bf16 v16bf;
typedef __attribute__((ext_vector_type(8)))  float  v8f;

#define WMMA_BF16(a, b, c) \
  __builtin_amdgcn_wmma_f32_16x16x32_bf16(false, (a), false, (b), (short)0, (c), false, false)

constexpr int Bc  = 16;
constexpr int Lc  = 512;
constexpr int Hc  = 256;
constexpr int NHc = 8;
constexpr int DHc = 32;    // Hc / NHc
constexpr int FFc = 1024;
constexpr int NLc = 2;
constexpr int RNc = 2 * Lc + 1;        // 1025 rows in the rel-pos table
constexpr int RNP = ((RNc + 15) / 16) * 16;   // 1040: padded rows for GEMM output

// ---------------------------------------------------------------------------
// Generic GEMM: C[M,N] = A[M,K] @ Bw[K,N] + bias[N], optional ReLU (template).
// One wave computes a 16x64 strip (4 adjacent 16x16 N-tiles): the A fragment
// is built once per K-step and feeds 4 WMMAs.  Out-of-range A rows are
// CLAMPED on load (an A row only feeds the same-numbered C row), and the C
// buffer is guaranteed padded to a multiple of 16 rows, so both the K-loop
// and the store epilogue are completely branch-free.
// ---------------------------------------------------------------------------
template <int RELU>
__global__ void txl_gemm_bf16wmma(const float* __restrict__ A,
                                  const float* __restrict__ Bw,
                                  const float* __restrict__ bias,
                                  float* __restrict__ C,
                                  int M, int N, int K)
{
  const int wave   = threadIdx.x >> 5;
  const int lane   = threadIdx.x & 31;
  const int lhalf  = lane >> 4;
  const int lq     = lane & 15;
  const int stripsN = N >> 6;                 // strips of 64 columns
  const int tilesM  = (M + 15) >> 4;
  const int strip = blockIdx.x * (blockDim.x >> 5) + wave;
  if (strip >= tilesM * stripsN) return;
  const int m0 = (strip / stripsN) << 4;
  const int n0 = (strip % stripsN) << 6;

  const int arow  = m0 + lq;
  const int arowc = (arow < M) ? arow : (M - 1);   // clamp: branch-free loads
  const float* Ap = A + (size_t)arowc * K;

  v8f accs[4];
  accs[0] = v8f{0.f,0.f,0.f,0.f,0.f,0.f,0.f,0.f};
  accs[1] = accs[0]; accs[2] = accs[0]; accs[3] = accs[0];

  for (int k0 = 0; k0 < K; k0 += 32) {
    // A fragment: lane = row (lq), K elems kb..kb+15, kb = lhalf*16
    v16bf af;
#pragma unroll
    for (int t = 0; t < 8; ++t) {
      const int koff = k0 + lhalf * 16 + 2 * t;
      float2 p = *reinterpret_cast<const float2*>(Ap + koff);
      af[2 * t]     = (__bf16)p.x;
      af[2 * t + 1] = (__bf16)p.y;
    }
    // B fragments: lane = col (lq), K elems kb2..kb2+15, kb2 = lhalf*16
    const float* Bp = Bw + (size_t)(k0 + lhalf * 16) * N + n0 + lq;
    v16bf b0, b1, b2, b3;
#pragma unroll
    for (int e = 0; e < 16; ++e) {
      const float* r = Bp + (size_t)e * N;
      b0[e] = (__bf16)r[0];
      b1[e] = (__bf16)r[16];
      b2[e] = (__bf16)r[32];
      b3[e] = (__bf16)r[48];
    }
    accs[0] = WMMA_BF16(af, b0, accs[0]);
    accs[1] = WMMA_BF16(af, b1, accs[1]);
    accs[2] = WMMA_BF16(af, b2, accs[2]);
    accs[3] = WMMA_BF16(af, b3, accs[3]);
  }

  // Branch-free epilogue: C is padded to tilesM*16 rows.
  float* Cp = C + (size_t)(m0 + lhalf * 8) * N + n0 + lq;
#pragma unroll
  for (int s = 0; s < 4; ++s) {
    const float bv = bias[n0 + s * 16 + lq];
#pragma unroll
    for (int v = 0; v < 8; ++v) {
      float r = accs[s][v] + bv;
      if (RELU) r = fmaxf(r, 0.f);
      Cp[(size_t)v * N + s * 16] = r;
    }
  }
}

// ---------------------------------------------------------------------------
// VR[n,r] = sum_d Rp[r, n*DH+d] * v[n,d]     (the "D" rank-1 term)
// ---------------------------------------------------------------------------
__global__ void txl_vrbias(const float* __restrict__ Rp,
                           const float* __restrict__ vvec,
                           float* __restrict__ VRout)
{
  const int idx = blockIdx.x * blockDim.x + threadIdx.x;
  if (idx >= NHc * RNc) return;
  const int nn = idx / RNc, r = idx % RNc;
  const float* rp = Rp + (size_t)r * Hc + nn * DHc;
  const float* vv = vvec + nn * DHc;
  float s = 0.f;
#pragma unroll
  for (int d = 0; d < DHc; ++d) s += rp[d] * vv[d];
  VRout[(size_t)nn * RNc + r] = s;
}

// ---------------------------------------------------------------------------
// Cb[b,n,j] = sum_d u[n,d] * K[b,j,n*DH+d]   (the "C" rank-1 term)
// ---------------------------------------------------------------------------
__global__ void txl_ubias(const float* __restrict__ Kb,
                          const float* __restrict__ uvec,
                          float* __restrict__ Cbout)
{
  const int idx = blockIdx.x * blockDim.x + threadIdx.x;
  if (idx >= Bc * NHc * Lc) return;
  const int j  = idx % Lc;
  const int nn = (idx / Lc) % NHc;
  const int bb = idx / (Lc * NHc);
  const float* kp = Kb + ((size_t)(bb * Lc + j) * Hc) + nn * DHc;
  const float* uu = uvec + nn * DHc;
  float s = 0.f;
#pragma unroll
  for (int d = 0; d < DHc; ++d) s += kp[d] * uu[d];
  Cbout[idx] = s;   // idx == (bb*NHc + nn)*Lc + j
}

// ---------------------------------------------------------------------------
// Fused rel-pos attention (flash-style, online softmax).
// Grid: (L/16, NH, B). Block: 32 threads (one wave) owning a 16-row q strip.
// Per 16-wide key tile:  S = q·k^T (1 WMMA)  +  gather of QRwin (2 WMMAs,
// Bt[i,j] = QRw[i, j-i+15] via LDS)  +  VR/Cb bias terms; then online softmax
// over 32-wide chunks and P·V (2 WMMAs) with P routed through LDS to convert
// C-layout -> A-fragment layout.
// ---------------------------------------------------------------------------
__global__ void txl_attn_wmma(const float* __restrict__ Q,
                              const float* __restrict__ Km,
                              const float* __restrict__ V,
                              const float* __restrict__ Rp,
                              const float* __restrict__ VR,
                              const float* __restrict__ Cb,
                              const int* __restrict__ seq_len,
                              float* __restrict__ O)
{
  __shared__ float  qrw[16][32];
  __shared__ float  sbuf[16][32];
  __shared__ __bf16 pbuf[16][32];
  __shared__ float  corrbuf[16];
  __shared__ float  rinvbuf[16];

  const int i0    = blockIdx.x << 4;
  const int n     = blockIdx.y;
  const int b     = blockIdx.z;
  const int lane  = threadIdx.x & 31;
  const int lhalf = lane >> 4;
  const int lq    = lane & 15;
  const int hoff  = n * DHc;
  const int slen  = seq_len[b];
  const float scale = 0.17677669529663687f;   // 1/sqrt(DH)
  const v8f z8 = {0.f, 0.f, 0.f, 0.f, 0.f, 0.f, 0.f, 0.f};

  // ---- Q fragment (A layout), kept in registers for the whole kernel ----
  const float* qrow = Q + ((size_t)(b * Lc + i0 + lq) * Hc) + hoff;
  v16bf aq;
#pragma unroll
  for (int t = 0; t < 8; ++t) {
    const int koff = lhalf * 16 + 2 * t;
    float2 p = *reinterpret_cast<const float2*>(qrow + koff);
    aq[2 * t]     = (__bf16)p.x;
    aq[2 * t + 1] = (__bf16)p.y;
  }

  float rm = -3.0e38f, rs = 0.f;          // row stats live in lanes 0..15
  v8f o0 = z8, o1 = z8;                   // 16x32 output accumulator

  for (int j0 = 0; j0 < Lc; j0 += 32) {
#pragma unroll
    for (int sub = 0; sub < 2; ++sub) {
      const int j0s = j0 + sub * 16;

      // K fragment: B[d, jj], lane jj = lq, d = lhalf*16 + e (contiguous)
      const float* kp = Km + ((size_t)(b * Lc + j0s + lq) * Hc) + hoff + lhalf * 16;
      const float4* k4 = reinterpret_cast<const float4*>(kp);
      v16bf bk;
#pragma unroll
      for (int q4 = 0; q4 < 4; ++q4) {
        float4 c = k4[q4];
        bk[4 * q4 + 0] = (__bf16)c.x; bk[4 * q4 + 1] = (__bf16)c.y;
        bk[4 * q4 + 2] = (__bf16)c.z; bk[4 * q4 + 3] = (__bf16)c.w;
      }
      v8f s = WMMA_BF16(aq, bk, z8);

      // Rel-pos window: rows rbase..rbase+31 of Rp; always in [0, RN)
      const int rbase = j0s - i0 + Lc - 15;
      const float* r0p = Rp + (size_t)(rbase + lq)      * Hc + hoff + lhalf * 16;
      const float* r1p = Rp + (size_t)(rbase + 16 + lq) * Hc + hoff + lhalf * 16;
      const float4* r04 = reinterpret_cast<const float4*>(r0p);
      const float4* r14 = reinterpret_cast<const float4*>(r1p);
      v16bf br0, br1;
#pragma unroll
      for (int q4 = 0; q4 < 4; ++q4) {
        float4 c0 = r04[q4], c1 = r14[q4];
        br0[4 * q4 + 0] = (__bf16)c0.x; br0[4 * q4 + 1] = (__bf16)c0.y;
        br0[4 * q4 + 2] = (__bf16)c0.z; br0[4 * q4 + 3] = (__bf16)c0.w;
        br1[4 * q4 + 0] = (__bf16)c1.x; br1[4 * q4 + 1] = (__bf16)c1.y;
        br1[4 * q4 + 2] = (__bf16)c1.z; br1[4 * q4 + 3] = (__bf16)c1.w;
      }
      v8f qr0 = WMMA_BF16(aq, br0, z8);
      v8f qr1 = WMMA_BF16(aq, br1, z8);

      __syncthreads();                      // previous consumers of qrw done
#pragma unroll
      for (int vv = 0; vv < 8; ++vv) {
        const int m = vv + lhalf * 8;
        qrw[m][lq]      = qr0[vv];
        qrw[m][16 + lq] = qr1[vv];
      }
      __syncthreads();

      // Assemble scores: A + Bt(shift-gather) + Ct + Dt, scale, mask
#pragma unroll
      for (int vv = 0; vv < 8; ++vv) {
        const int m = vv + lhalf * 8;
        const int j = j0s + lq;
        float sc = s[vv]
                 + qrw[m][lq - m + 15]
                 + VR[(size_t)n * RNc + (j - (i0 + m) + Lc)]
                 + Cb[((size_t)(b * NHc) + n) * Lc + j];
        sc *= scale;
        if (j >= slen) sc = -1e15f;
        sbuf[m][sub * 16 + lq] = sc;
      }
    }
    __syncthreads();

    // ---- online softmax for this 32-wide chunk (lanes 0..15 own rows) ----
    if (lane < 16) {
      float cmax = -3.0e38f;
#pragma unroll
      for (int c = 0; c < 32; ++c) cmax = fmaxf(cmax, sbuf[lq][c]);
      const float nm   = fmaxf(rm, cmax);
      const float corr = __expf(rm - nm);
      float csum = 0.f;
#pragma unroll
      for (int c = 0; c < 32; ++c) {
        const float p = __expf(sbuf[lq][c] - nm);
        csum += p;
        pbuf[lq][c] = (__bf16)p;
      }
      rs = rs * corr + csum;
      rm = nm;
      corrbuf[lq] = corr;
    }
    __syncthreads();

    // rescale output accumulators by per-row correction
#pragma unroll
    for (int vv = 0; vv < 8; ++vv) {
      const float c = corrbuf[vv + lhalf * 8];
      o0[vv] *= c;
      o1[vv] *= c;
    }

    // P fragment (A layout) from LDS
    v16bf ap;
#pragma unroll
    for (int e = 0; e < 16; ++e) {
      const int kk = lhalf * 16 + e;
      ap[e] = pbuf[lq][kk];
    }
    // V fragments: cols d = lq / 16+lq, K elems kk = lhalf*16 + e (strided)
    const float* vb = V + ((size_t)(b * Lc + j0 + lhalf * 16) * Hc) + hoff;
    v16bf bv0, bv1;
#pragma unroll
    for (int e = 0; e < 16; ++e) {
      bv0[e] = (__bf16)vb[(size_t)e * Hc + lq];
      bv1[e] = (__bf16)vb[(size_t)e * Hc + 16 + lq];
    }
    o0 = WMMA_BF16(ap, bv0, o0);
    o1 = WMMA_BF16(ap, bv1, o1);
    __syncthreads();
  }

  if (lane < 16) rinvbuf[lq] = 1.0f / rs;
  __syncthreads();

  float* ob = O + ((size_t)(b * Lc + i0) * Hc) + hoff;
#pragma unroll
  for (int vv = 0; vv < 8; ++vv) {
    const int m = vv + lhalf * 8;
    const float inv = rinvbuf[m];
    ob[(size_t)m * Hc + lq]      = o0[vv] * inv;
    ob[(size_t)m * Hc + 16 + lq] = o1[vv] * inv;
  }
}

// ---------------------------------------------------------------------------
// out = LayerNorm(2*in) * g + b   (matches reference _ln(o + o))
// One 256-thread block per row of H=256.
// ---------------------------------------------------------------------------
__global__ void txl_ln2x(const float* __restrict__ in,
                         const float* __restrict__ g,
                         const float* __restrict__ bta,
                         float* __restrict__ out)
{
  const int row = blockIdx.x;
  const int t   = threadIdx.x;
  float x = 2.0f * in[(size_t)row * Hc + t];

  __shared__ float red[8];
  __shared__ float stat[2];

  float s = x;
#pragma unroll
  for (int off = 16; off > 0; off >>= 1) s += __shfl_down(s, off, 32);
  if ((t & 31) == 0) red[t >> 5] = s;
  __syncthreads();
  if (t == 0) {
    float a = 0.f;
    for (int i = 0; i < 8; ++i) a += red[i];
    stat[0] = a * (1.0f / Hc);
  }
  __syncthreads();
  const float m = stat[0];
  const float d = x - m;

  float s2 = d * d;
#pragma unroll
  for (int off = 16; off > 0; off >>= 1) s2 += __shfl_down(s2, off, 32);
  __syncthreads();
  if ((t & 31) == 0) red[t >> 5] = s2;
  __syncthreads();
  if (t == 0) {
    float a = 0.f;
    for (int i = 0; i < 8; ++i) a += red[i];
    stat[1] = a * (1.0f / Hc);
  }
  __syncthreads();
  const float inv = rsqrtf(stat[1] + 1e-5f);
  out[(size_t)row * Hc + t] = d * inv * g[t] + bta[t];
}

// ---------------------------------------------------------------------------
// Host-side orchestration
// ---------------------------------------------------------------------------
extern "C" void kernel_launch(void* const* d_in, const int* in_sizes, int n_in,
                              void* d_out, int out_size, void* d_ws, size_t ws_size,
                              hipStream_t stream)
{
  (void)in_sizes; (void)n_in; (void)out_size; (void)ws_size;

  const float* x   = (const float*)d_in[0];
  const int*   sl  = (const int*)  d_in[1];
  const float* pe  = (const float*)d_in[2];
  const float* Wq  = (const float*)d_in[3];
  const float* bq  = (const float*)d_in[4];
  const float* Wk  = (const float*)d_in[5];
  const float* bk  = (const float*)d_in[6];
  const float* Wv  = (const float*)d_in[7];
  const float* bv  = (const float*)d_in[8];
  const float* Wr  = (const float*)d_in[9];
  const float* br  = (const float*)d_in[10];
  const float* u   = (const float*)d_in[11];
  const float* vv  = (const float*)d_in[12];
  const float* Wo  = (const float*)d_in[13];
  const float* bo  = (const float*)d_in[14];
  const float* g1  = (const float*)d_in[15];
  const float* b1l = (const float*)d_in[16];
  const float* W1  = (const float*)d_in[17];
  const float* b1f = (const float*)d_in[18];
  const float* W2  = (const float*)d_in[19];
  const float* b2f = (const float*)d_in[20];
  const float* g2  = (const float*)d_in[21];
  const float* b2l = (const float*)d_in[22];

  float* ws = (float*)d_ws;
  const size_t BLH = (size_t)Bc * Lc * Hc;
  float* Xb  = ws;                                 // current activations
  float* Qb  = Xb  + BLH;                          // Q   / later T1 (post-Wo)
  float* Kb  = Qb  + BLH;                          // K   / later h  (post-LN1)
  float* Vb  = Kb  + BLH;                          // V   / later f  (post-W2)
  float* Ob  = Vb  + BLH;                          // attention output
  float* Rpb = Ob  + BLH;                          // projected rel-pos table (padded rows)
  float* VRb = Rpb + (size_t)RNP * Hc;             // VR[n, r]
  float* Cbb = VRb + (size_t)NHc * RNc;            // Cb[b, n, j]
  float* FFb = Cbb + (size_t)Bc * NHc * Lc;        // FFN hidden

  hipMemcpyAsync(Xb, x, BLH * sizeof(float), hipMemcpyDeviceToDevice, stream);

  auto gemm = [&](const float* A, const float* Bw, const float* bias, float* C,
                  int M, int N, int K, int relu) {
    const int strips = ((M + 15) / 16) * (N / 64);   // 16x64 per wave
    const int blocks = (strips + 3) / 4;             // 4 waves per block
    if (relu)
      txl_gemm_bf16wmma<1><<<blocks, 128, 0, stream>>>(A, Bw, bias, C, M, N, K);
    else
      txl_gemm_bf16wmma<0><<<blocks, 128, 0, stream>>>(A, Bw, bias, C, M, N, K);
  };

  const int Mrows = Bc * Lc;

  for (int l = 0; l < NLc; ++l) {
    const size_t HH = (size_t)Hc * Hc;
    gemm(Xb, Wq + l * HH, bq + l * Hc, Qb, Mrows, Hc, Hc, 0);
    gemm(Xb, Wk + l * HH, bk + l * Hc, Kb, Mrows, Hc, Hc, 0);
    gemm(Xb, Wv + l * HH, bv + l * Hc, Vb, Mrows, Hc, Hc, 0);
    gemm(pe, Wr + l * HH, br + l * Hc, Rpb, RNc, Hc, Hc, 0);  // rows padded to RNP

    {
      const int n1 = NHc * RNc;
      txl_vrbias<<<(n1 + 255) / 256, 256, 0, stream>>>(
          Rpb, vv + (size_t)l * NHc * DHc, VRb);
      const int n2 = Bc * NHc * Lc;
      txl_ubias<<<(n2 + 255) / 256, 256, 0, stream>>>(
          Kb, u + (size_t)l * NHc * DHc, Cbb);
    }

    txl_attn_wmma<<<dim3(Lc / 16, NHc, Bc), 32, 0, stream>>>(
        Qb, Kb, Vb, Rpb, VRb, Cbb, sl, Ob);

    gemm(Ob, Wo + l * HH, bo + l * Hc, Qb, Mrows, Hc, Hc, 0);      // T1 -> Qb
    txl_ln2x<<<Mrows, Hc, 0, stream>>>(Qb, g1 + l * Hc, b1l + l * Hc, Kb); // h -> Kb
    gemm(Kb, W1 + (size_t)l * Hc * FFc, b1f + l * FFc, FFb, Mrows, FFc, Hc, 1);
    gemm(FFb, W2 + (size_t)l * FFc * Hc, b2f + l * Hc, Vb, Mrows, Hc, FFc, 0);

    float* outp = (l == NLc - 1) ? (float*)d_out : Xb;
    txl_ln2x<<<Mrows, Hc, 0, stream>>>(Vb, g2 + l * Hc, b2l + l * Hc, outp);
  }
}
